// EnhancedMultiHeadAttention_20521353740566
// MI455X (gfx1250) — compile-verified, matrix operand order checked by reading, not run on hardware
//
#include <hip/hip_runtime.h>
#include <cstdint>
#include <cstddef>

// ---------------------------------------------------------------------------
// Problem constants (match the JAX reference)
// ---------------------------------------------------------------------------
#define B_   256
#define L_   64
#define D_   1024
#define H_   16
#define HD_  64          // local head dim
#define HG_  8
#define HDG_ 128         // global head dim
#define BL_  (B_ * L_)   // 16384 rows
#define DD_  (D_ * D_)   // 1M weight elems
#define RELPAD_ 80       // 65 real rel rows padded to 80 (multiple of 16)

typedef __attribute__((ext_vector_type(16))) __bf16 v16bf;
typedef __attribute__((ext_vector_type(8)))  __bf16 v8bf;
typedef __attribute__((ext_vector_type(8)))  float  v8f;

// ---------------------------------------------------------------------------
// Fragment loader: 16x32 bf16 tile, one row per lane-pair, contiguous K.
// Lanes 0-15  -> rows r0..r0+15, K chunks [k0+0..7]   and [k0+16..23]
// Lanes 16-31 -> rows r0..r0+15, K chunks [k0+8..15]  and [k0+24..31]
// ---------------------------------------------------------------------------
__device__ __forceinline__ v16bf load_frag(const __bf16* __restrict__ p,
                                           int ld, int r0, int k0) {
  const int lane = threadIdx.x & 31;
  const __bf16* row = p + (size_t)(r0 + (lane & 15)) * ld + k0 + ((lane >> 4) << 3);
  v8bf lo = *reinterpret_cast<const v8bf*>(row);
  v8bf hi = *reinterpret_cast<const v8bf*>(row + 16);
  v16bf f;
#pragma unroll
  for (int i = 0; i < 8; ++i) { f[i] = lo[i]; f[i + 8] = hi[i]; }
  return f;
}

__device__ __forceinline__ v8f wmma_bf16(v16bf a, v16bf b, v8f c) {
  return __builtin_amdgcn_wmma_f32_16x16x32_bf16(
      /*neg_a=*/false, a, /*neg_b=*/false, b,
      /*c_mod=*/(short)0, c, /*reuse_a=*/false, /*reuse_b=*/false);
}

// Async DMA: one 16-byte chunk per lane, global -> LDS, tracked by ASYNCcnt.
// Low 32 bits of a flat shared-memory pointer are the wave-relative LDS byte
// address (flat-LDS encoding keeps the offset in addr[31:0]).
__device__ __forceinline__ void async_cp16(void* lds_ptr,
                                           const __bf16* gptr) {
  const unsigned lds_addr = (unsigned)(uintptr_t)lds_ptr;
  asm volatile("global_load_async_to_lds_b128 %0, %1, off"
               :: "v"(lds_addr), "v"(gptr)
               : "memory");
}

__device__ __forceinline__ void wait_async_le6() {
  asm volatile("s_wait_asynccnt 0x6" ::: "memory");
}
__device__ __forceinline__ void wait_async_0() {
  asm volatile("s_wait_asynccnt 0x0" ::: "memory");
}

// ---------------------------------------------------------------------------
// fp32 -> bf16 conversion (grid-stride)
// ---------------------------------------------------------------------------
__global__ void cvt_bf16_kernel(const float* __restrict__ in,
                                __bf16* __restrict__ out, int n) {
  for (int i = blockIdx.x * blockDim.x + threadIdx.x; i < n;
       i += gridDim.x * blockDim.x)
    out[i] = (__bf16)in[i];
}

// rel_k (65 x 64) -> bf16 padded to (80 x 64) with zero rows
__global__ void pad_relk_kernel(const float* __restrict__ relk,
                                __bf16* __restrict__ out) {
  int i = blockIdx.x * blockDim.x + threadIdx.x;
  if (i < RELPAD_ * HD_) {
    int j = i >> 6;  // row
    out[i] = (j < 65) ? (__bf16)relk[i] : (__bf16)0.0f;
  }
}

// combined = 0.7*local + 0.3*glob  (bf16 in, bf16 out)
__global__ void combine_kernel(const __bf16* __restrict__ a,
                               const __bf16* __restrict__ b,
                               __bf16* __restrict__ o, int n) {
  for (int i = blockIdx.x * blockDim.x + threadIdx.x; i < n;
       i += gridDim.x * blockDim.x)
    o[i] = (__bf16)(0.7f * (float)a[i] + 0.3f * (float)b[i]);
}

// ---------------------------------------------------------------------------
// C = A @ W^T + bias.  A: [M,K] bf16 rm.  W: [N,K] bf16 rm.
// 256 threads = 8 waves (2x4 grid); block tile 128(M) x 256(N); each wave
// owns a 64x64 C tile (4x4 wmma accumulators).  A/W K-slabs are staged in a
// double-buffered LDS pipeline fed by GLOBAL_LOAD_ASYNC_TO_LDS_B128: each
// thread issues exactly 6 chunks/stage (2 A + 4 W) so every wave has exactly
// 6 ASYNCcnt entries per stage -> s_wait_asynccnt 6 retires the previous
// stage while the next streams in; s_barrier publishes it across waves.
// ---------------------------------------------------------------------------
template <bool OUT_F32>
__global__ __launch_bounds__(256) void gemm_xwT_kernel(
    const __bf16* __restrict__ A, const __bf16* __restrict__ W,
    const float* __restrict__ bias, void* __restrict__ Cout,
    int M, int N, int K) {
  __shared__ alignas(16) __bf16 As[2][128][32];
  __shared__ alignas(16) __bf16 Ws[2][256][32];

  const int tid = threadIdx.x;
  const int lane = tid & 31;
  const int wave = tid >> 5;
  const int mblk = blockIdx.y * 128;
  const int nblk = blockIdx.x * 256;
  const int wm = (wave >> 2) * 64;  // wave row (2)
  const int wn = (wave & 3) * 64;   // wave col (4)

  auto issue_stage = [&](int st, int k0) {
#pragma unroll
    for (int i = 0; i < 2; ++i) {                 // A: 128x32 = 512 chunks
      const int c = tid + 256 * i;
      const int r = c >> 2, c8 = (c & 3) << 3;
      async_cp16(&As[st][r][c8], A + (size_t)(mblk + r) * K + k0 + c8);
    }
#pragma unroll
    for (int i = 0; i < 4; ++i) {                 // W: 256x32 = 1024 chunks
      const int c = tid + 256 * i;
      const int r = c >> 2, c8 = (c & 3) << 3;
      async_cp16(&Ws[st][r][c8], W + (size_t)(nblk + r) * K + k0 + c8);
    }
  };

  v8f acc[4][4];
#pragma unroll
  for (int i = 0; i < 4; ++i)
#pragma unroll
    for (int j = 0; j < 4; ++j) acc[i][j] = {};

  const int NK = K / 32;
  issue_stage(0, 0);
  for (int kt = 0; kt < NK; ++kt) {
    const int cur = kt & 1;
    if (kt + 1 < NK) {
      issue_stage((kt + 1) & 1, (kt + 1) * 32);
      wait_async_le6();   // stage kt complete; stage kt+1 in flight
    } else {
      wait_async_0();
    }
    __syncthreads();      // publish stage kt across all 8 waves

    v16bf a[4], b[4];
#pragma unroll
    for (int i = 0; i < 4; ++i) a[i] = load_frag(&As[cur][0][0], 32, wm + 16 * i, 0);
#pragma unroll
    for (int j = 0; j < 4; ++j) b[j] = load_frag(&Ws[cur][0][0], 32, wn + 16 * j, 0);
#pragma unroll
    for (int i = 0; i < 4; ++i)
#pragma unroll
      for (int j = 0; j < 4; ++j) acc[i][j] = wmma_bf16(a[i], b[j], acc[i][j]);

    __syncthreads();      // protect buffer `cur` from the stage kt+2 overwrite
  }

  const int rhi = (lane >> 4) << 3;   // +8 rows for upper half-wave
  const int col = lane & 15;
#pragma unroll
  for (int i = 0; i < 4; ++i)
#pragma unroll
    for (int j = 0; j < 4; ++j) {
      const int c = nblk + wn + 16 * j + col;
      const float bb = bias[c];
#pragma unroll
      for (int g = 0; g < 8; ++g) {
        const int r = mblk + wm + 16 * i + rhi + g;
        const float v = acc[i][j][g] + bb;
        if (OUT_F32)
          ((float*)Cout)[(size_t)r * N + c] = v;
        else
          ((__bf16*)Cout)[(size_t)r * N + c] = (__bf16)v;
      }
    }
}

// ---------------------------------------------------------------------------
// Local attention, one block per (b,h).  hd = 64.
//   S = q k^T / 8 ;  S += gather(q @ rel_pad^T) ;  P = softmax(S) ; O = P v
// Each of the 4 waves owns 16 query rows end-to-end (no cross-wave sync
// needed after the initial load barrier; same-wave LDS ops are in-order).
// ---------------------------------------------------------------------------
__global__ __launch_bounds__(128) void local_attn_kernel(
    const __bf16* __restrict__ xq, const __bf16* __restrict__ xk,
    const __bf16* __restrict__ xv, const __bf16* __restrict__ relk,
    __bf16* __restrict__ out) {
  __shared__ alignas(16) __bf16 sq[64][64];
  __shared__ alignas(16) __bf16 sk[64][64];
  __shared__ alignas(16) __bf16 svT[64][64];   // transposed V: [d][m]
  __shared__ alignas(16) __bf16 sp[64][64];    // softmax(S) in bf16
  __shared__ float S[64][72];
  __shared__ float Pq[64][80];

  const int b = blockIdx.x >> 4;   // H = 16
  const int h = blockIdx.x & 15;
  const size_t base = (size_t)b * L_ * D_ + (size_t)h * HD_;

  for (int idx = threadIdx.x; idx < 64 * 8; idx += 128) {
    const int r = idx >> 3, c = (idx & 7) << 3;
    *(v8bf*)&sq[r][c] = *(const v8bf*)&xq[base + (size_t)r * D_ + c];
    *(v8bf*)&sk[r][c] = *(const v8bf*)&xk[base + (size_t)r * D_ + c];
    v8bf vv = *(const v8bf*)&xv[base + (size_t)r * D_ + c];
#pragma unroll
    for (int t = 0; t < 8; ++t) svT[c + t][r] = vv[t];
  }
  __syncthreads();

  const int wave = threadIdx.x >> 5;
  const int lane = threadIdx.x & 31;
  const int m0 = wave * 16;
  const int rhi = (lane >> 4) << 3;
  const int col = lane & 15;

  // ---- S = q k^T * (1/8) -------------------------------------------------
#pragma unroll
  for (int j = 0; j < 4; ++j) {
    v8f acc = {};
#pragma unroll
    for (int kk = 0; kk < 64; kk += 32)
      acc = wmma_bf16(load_frag(&sq[0][0], 64, m0, kk),
                      load_frag(&sk[0][0], 64, j * 16, kk), acc);
#pragma unroll
    for (int g = 0; g < 8; ++g) S[m0 + rhi + g][j * 16 + col] = acc[g] * 0.125f;
  }

  // ---- Pq = q @ rel_pad^T  (16 x 80 strip per wave) ----------------------
#pragma unroll
  for (int j = 0; j < 5; ++j) {
    v8f acc = {};
#pragma unroll
    for (int kk = 0; kk < 64; kk += 32)
      acc = wmma_bf16(load_frag(&sq[0][0], 64, m0, kk),
                      load_frag(relk, 64, j * 16, kk), acc);
#pragma unroll
    for (int g = 0; g < 8; ++g) Pq[m0 + rhi + g][j * 16 + col] = acc[g];
  }

  // ---- bias gather + softmax (one lane per query row) --------------------
  if (lane < 16) {
    const int l = m0 + lane;
    float mx = -3.4e38f;
    for (int r = 0; r < 64; ++r) {
      int idx = r - l + 32;
      idx = idx < 0 ? 0 : (idx > 64 ? 64 : idx);
      const float s = S[l][r] + Pq[l][idx];
      S[l][r] = s;
      mx = fmaxf(mx, s);
    }
    float sum = 0.0f;
    for (int r = 0; r < 64; ++r) {
      const float e = __expf(S[l][r] - mx);
      S[l][r] = e;
      sum += e;
    }
    const float inv = 1.0f / sum;
    for (int r = 0; r < 64; ++r) sp[l][r] = (__bf16)(S[l][r] * inv);
  }

  // ---- O = P @ V  (B fragments are rows of transposed V) -----------------
#pragma unroll
  for (int j = 0; j < 4; ++j) {
    v8f acc = {};
#pragma unroll
    for (int kk = 0; kk < 64; kk += 32)
      acc = wmma_bf16(load_frag(&sp[0][0], 64, m0, kk),
                      load_frag(&svT[0][0], 64, j * 16, kk), acc);
    const int c = j * 16 + col;
#pragma unroll
    for (int g = 0; g < 8; ++g)
      out[base + (size_t)(m0 + rhi + g) * D_ + c] = (__bf16)acc[g];
  }
}

// ---------------------------------------------------------------------------
// Global attention, one block per (b,hg).  hd = 128, no bias.
// ---------------------------------------------------------------------------
__global__ __launch_bounds__(128) void global_attn_kernel(
    const __bf16* __restrict__ gq, const __bf16* __restrict__ gk,
    const __bf16* __restrict__ gv, __bf16* __restrict__ out) {
  __shared__ alignas(16) __bf16 sq[64][128];
  __shared__ alignas(16) __bf16 sk[64][128];
  __shared__ alignas(16) __bf16 svT[128][64];
  __shared__ alignas(16) __bf16 sp[64][64];
  __shared__ float S[64][72];

  const int b = blockIdx.x >> 3;   // HG = 8
  const int hg = blockIdx.x & 7;
  const size_t base = (size_t)b * L_ * D_ + (size_t)hg * HDG_;

  for (int idx = threadIdx.x; idx < 64 * 16; idx += 128) {
    const int r = idx >> 4, c = (idx & 15) << 3;
    *(v8bf*)&sq[r][c] = *(const v8bf*)&gq[base + (size_t)r * D_ + c];
    *(v8bf*)&sk[r][c] = *(const v8bf*)&gk[base + (size_t)r * D_ + c];
    v8bf vv = *(const v8bf*)&gv[base + (size_t)r * D_ + c];
#pragma unroll
    for (int t = 0; t < 8; ++t) svT[c + t][r] = vv[t];
  }
  __syncthreads();

  const int wave = threadIdx.x >> 5;
  const int lane = threadIdx.x & 31;
  const int m0 = wave * 16;
  const int rhi = (lane >> 4) << 3;
  const int col = lane & 15;
  const float scale = 0.08838834764831845f;  // 1/sqrt(128)

#pragma unroll
  for (int j = 0; j < 4; ++j) {
    v8f acc = {};
#pragma unroll
    for (int kk = 0; kk < 128; kk += 32)
      acc = wmma_bf16(load_frag(&sq[0][0], 128, m0, kk),
                      load_frag(&sk[0][0], 128, j * 16, kk), acc);
#pragma unroll
    for (int g = 0; g < 8; ++g) S[m0 + rhi + g][j * 16 + col] = acc[g] * scale;
  }

  if (lane < 16) {
    const int l = m0 + lane;
    float mx = -3.4e38f;
    for (int r = 0; r < 64; ++r) mx = fmaxf(mx, S[l][r]);
    float sum = 0.0f;
    for (int r = 0; r < 64; ++r) {
      const float e = __expf(S[l][r] - mx);
      S[l][r] = e;
      sum += e;
    }
    const float inv = 1.0f / sum;
    for (int r = 0; r < 64; ++r) sp[l][r] = (__bf16)(S[l][r] * inv);
  }

#pragma unroll
  for (int j = 0; j < 8; ++j) {       // 128 output dims
    v8f acc = {};
#pragma unroll
    for (int kk = 0; kk < 64; kk += 32)
      acc = wmma_bf16(load_frag(&sp[0][0], 64, m0, kk),
                      load_frag(&svT[0][0], 64, j * 16, kk), acc);
    const int c = j * 16 + col;
#pragma unroll
    for (int g = 0; g < 8; ++g)
      out[base + (size_t)(m0 + rhi + g) * D_ + c] = (__bf16)acc[g];
  }
}

// ---------------------------------------------------------------------------
// Host-side orchestration
// ---------------------------------------------------------------------------
extern "C" void kernel_launch(void* const* d_in, const int* in_sizes, int n_in,
                              void* d_out, int out_size, void* d_ws,
                              size_t ws_size, hipStream_t stream) {
  (void)in_sizes; (void)n_in; (void)out_size; (void)ws_size;

  const float* query   = (const float*)d_in[0];
  const float* key     = (const float*)d_in[1];
  const float* value   = (const float*)d_in[2];
  const float* Wq      = (const float*)d_in[3];
  const float* bq      = (const float*)d_in[4];
  const float* Wk      = (const float*)d_in[5];
  const float* bk      = (const float*)d_in[6];
  const float* Wv      = (const float*)d_in[7];
  const float* bv      = (const float*)d_in[8];
  const float* Wo      = (const float*)d_in[9];
  const float* bo      = (const float*)d_in[10];
  const float* rel_k   = (const float*)d_in[11];
  // d_in[12] = rel_v : looked up but unused upstream in the reference
  const float* g_in_w  = (const float*)d_in[13];
  const float* g_in_b  = (const float*)d_in[14];
  const float* g_out_w = (const float*)d_in[15];
  const float* g_out_b = (const float*)d_in[16];

  char* ws = (char*)d_ws;
  size_t off = 0;
  auto alloc = [&](size_t bytes) -> char* {
    char* p = ws + off;
    off += (bytes + 255) & ~(size_t)255;
    return p;
  };

  const size_t BLD = (size_t)BL_ * D_;

  __bf16* qbf   = (__bf16*)alloc(BLD * 2);
  __bf16* kbf   = (__bf16*)alloc(BLD * 2);
  __bf16* vbf   = (__bf16*)alloc(BLD * 2);
  __bf16* wqbf  = (__bf16*)alloc((size_t)DD_ * 2);
  __bf16* wkbf  = (__bf16*)alloc((size_t)DD_ * 2);
  __bf16* wvbf  = (__bf16*)alloc((size_t)DD_ * 2);
  __bf16* wobf  = (__bf16*)alloc((size_t)DD_ * 2);
  __bf16* ginbf = (__bf16*)alloc((size_t)3 * DD_ * 2);
  __bf16* goutbf= (__bf16*)alloc((size_t)DD_ * 2);
  __bf16* relbf = (__bf16*)alloc((size_t)RELPAD_ * HD_ * 2);
  __bf16* xq    = (__bf16*)alloc(BLD * 2);
  __bf16* xk    = (__bf16*)alloc(BLD * 2);
  __bf16* xv    = (__bf16*)alloc(BLD * 2);
  __bf16* gq    = (__bf16*)alloc(BLD * 2);
  __bf16* gk    = (__bf16*)alloc(BLD * 2);
  __bf16* gv    = (__bf16*)alloc(BLD * 2);
  // After the six projection GEMMs, the bf16 activation inputs are dead:
  // alias the attention outputs / combined buffer onto them.
  __bf16* loc  = qbf;   // local attention output  [BL, D]
  __bf16* gat  = kbf;   // global attention output [BL, D]
  __bf16* gpp  = vbf;   // global out-projection   [BL, D]
  __bf16* comb = xq;    // 0.7*loc + 0.3*gpp       [BL, D]

  // ---- fp32 -> bf16 conversions -----------------------------------------
  const int cvtN = (int)BLD;
  cvt_bf16_kernel<<<4096, 256, 0, stream>>>(query, qbf, cvtN);
  cvt_bf16_kernel<<<4096, 256, 0, stream>>>(key,   kbf, cvtN);
  cvt_bf16_kernel<<<4096, 256, 0, stream>>>(value, vbf, cvtN);
  cvt_bf16_kernel<<<1024, 256, 0, stream>>>(Wq, wqbf, DD_);
  cvt_bf16_kernel<<<1024, 256, 0, stream>>>(Wk, wkbf, DD_);
  cvt_bf16_kernel<<<1024, 256, 0, stream>>>(Wv, wvbf, DD_);
  cvt_bf16_kernel<<<1024, 256, 0, stream>>>(Wo, wobf, DD_);
  cvt_bf16_kernel<<<3072, 256, 0, stream>>>(g_in_w, ginbf, 3 * DD_);
  cvt_bf16_kernel<<<1024, 256, 0, stream>>>(g_out_w, goutbf, DD_);
  pad_relk_kernel<<<(RELPAD_ * HD_ + 255) / 256, 256, 0, stream>>>(rel_k, relbf);

  // ---- six input-projection GEMMs ---------------------------------------
  const dim3 gg(D_ / 256, BL_ / 128);  // (4, 128)
  gemm_xwT_kernel<false><<<gg, 256, 0, stream>>>(qbf, wqbf, bq, xq, BL_, D_, D_);
  gemm_xwT_kernel<false><<<gg, 256, 0, stream>>>(kbf, wkbf, bk, xk, BL_, D_, D_);
  gemm_xwT_kernel<false><<<gg, 256, 0, stream>>>(vbf, wvbf, bv, xv, BL_, D_, D_);
  gemm_xwT_kernel<false><<<gg, 256, 0, stream>>>(qbf, ginbf + 0 * (size_t)DD_,
                                                 g_in_b + 0 * D_, gq, BL_, D_, D_);
  gemm_xwT_kernel<false><<<gg, 256, 0, stream>>>(kbf, ginbf + 1 * (size_t)DD_,
                                                 g_in_b + 1 * D_, gk, BL_, D_, D_);
  gemm_xwT_kernel<false><<<gg, 256, 0, stream>>>(vbf, ginbf + 2 * (size_t)DD_,
                                                 g_in_b + 2 * D_, gv, BL_, D_, D_);

  // ---- attention branches ------------------------------------------------
  local_attn_kernel<<<B_ * H_, 128, 0, stream>>>(xq, xk, xv, relbf, loc);
  global_attn_kernel<<<B_ * HG_, 128, 0, stream>>>(gq, gk, gv, gat);

  // ---- global out-projection, combine, final projection ------------------
  gemm_xwT_kernel<false><<<gg, 256, 0, stream>>>(gat, goutbf, g_out_b, gpp,
                                                 BL_, D_, D_);
  combine_kernel<<<4096, 256, 0, stream>>>(loc, gpp, comb, cvtN);
  gemm_xwT_kernel<true><<<gg, 256, 0, stream>>>(comb, wobf, bo, d_out,
                                                BL_, D_, D_);
}
